// PyramidMambaFusionBlock_44744969290138
// MI455X (gfx1250) — compile-verified, matrix operand order checked by reading, not run on hardware
//
#include <hip/hip_runtime.h>
#include <math.h>

typedef unsigned short u16;
typedef __attribute__((ext_vector_type(16))) __bf16 v16bf;
typedef __attribute__((ext_vector_type(8)))  float  v8f;

#define DEV __device__ __forceinline__

static constexpr int BB   = 2;      // batch
static constexpr int NTOK = 4000;
static constexpr int CM   = 256;    // d_model
static constexpr int DI   = 512;    // d_inner
static constexpr int H0   = 64;
static constexpr int L0   = 4096;
static constexpr int KD   = 4;      // scan directions
static constexpr int RNK  = 16;     // dt rank
static constexpr int NST  = 16;     // ssm state

// CDNA5 async global->LDS path (guarded: falls back to manual staging)
#if defined(__gfx1250__) && __has_builtin(__builtin_amdgcn_global_load_async_to_lds_b128) && \
    __has_builtin(__builtin_amdgcn_global_load_async_to_lds_b64) && \
    __has_builtin(__builtin_amdgcn_s_wait_asynccnt)
#define HAVE_ASYNC_LDS 1
#define AS1 __attribute__((address_space(1)))
#define AS3 __attribute__((address_space(3)))
typedef int v4i __attribute__((vector_size(16)));
typedef int v2i __attribute__((vector_size(8)));
#endif

DEV u16 f2bf(float f) {
    unsigned x = __float_as_uint(f);
    unsigned r = x + 0x7fffu + ((x >> 16) & 1u);
    return (u16)(r >> 16);
}
DEV float bf2f(u16 s) { return __uint_as_float(((unsigned)s) << 16); }
DEV float siluf(float x) { return x / (1.f + __expf(-x)); }
DEV float softplusf(float x) { return (x > 20.f) ? x : log1pf(__expf(x)); }

union FragBF { v16bf v; unsigned u[8]; };

// ---------------------------------------------------------------------------
// Generic strided-batched bf16 GEMM: C[M,N] = A[M,K] * B[K,N] (+bias over N)
// Block tile 64x128, BK=32, 8 waves in 2x4 grid, wave tile 32x32:
// 2 A-frags x 2 B-frags -> 4 v_wmma_f32_16x16x32_bf16 per K-step.
// A tile staged via async global->LDS; both LDS tiles use 40-halfword (80B)
// row stride: 16B-aligned so fragment gathers lower to ds_load_b128, and the
// 20-bank stride is conflict-free across the 16 rows a fragment reads.
// B tile loaded to VGPRs and stored transposed (Bs[n][k]).
// Requires: K%32==0, N%128==0, lda/ldb%8==0. M guarded (x_proj M=48 case).
// ---------------------------------------------------------------------------
static constexpr int ASTR = 40;   // As row stride (u16)
static constexpr int BSTR = 40;   // Bs row stride (u16)

DEV void load_frag(const u16* Sm, int stride, int r0, int lane, FragBF& f) {
    const u16* row = Sm + (r0 + (lane & 15)) * stride;
    const int g = (lane >> 4) << 3;
#pragma unroll
    for (int j = 0; j < 4; ++j) {
        f.u[j]     = *(const unsigned*)(row + g + 2 * j);
        f.u[4 + j] = *(const unsigned*)(row + 16 + g + 2 * j);
    }
}

__global__ __launch_bounds__(256)
void gemm_bf16_kernel(const u16* __restrict__ A, const u16* __restrict__ Bm,
                      float* __restrict__ C, const float* __restrict__ bias,
                      int M, int N, int K, int lda, int ldb, int ldc,
                      long long sA, long long sB, long long sC, int aMod)
{
    __shared__ u16 As[64 * ASTR];
    __shared__ u16 Bs[128 * BSTR];   // transposed: Bs[n][k]
    const int z = blockIdx.z;
    A  += (long long)(z % aMod) * sA;
    Bm += (long long)z * sB;
    C  += (long long)z * sC;
    const int m0 = blockIdx.y * 64;
    const int n0 = blockIdx.x * 128;
    const int tid  = threadIdx.x;
    const int lane = tid & 31, wid = tid >> 5;
    const int wm = (wid & 1) << 5;   // wave row band (2 x 32 = 64)
    const int wn = (wid >> 1) << 5;  // wave col band (4 x 32 = 128)
    v8f a00 = {0.f,0.f,0.f,0.f,0.f,0.f,0.f,0.f};
    v8f a01 = a00, a10 = a00, a11 = a00;
    // A tile: 64 rows x 32 cols, one 16B chunk per thread
    const int ar = tid >> 2, ac = (tid & 3) << 3;
    const int arow = m0 + ar;
    // B tile: 32 rows x 128 cols, 16 elements (2 x uint4) per thread
    const int bk = tid >> 3, bn = (tid & 7) << 4;

    for (int k0 = 0; k0 < K; k0 += 32) {
        uint4 bv0 = *(const uint4*)(Bm + (long long)(k0 + bk) * ldb + n0 + bn);
        uint4 bv1 = *(const uint4*)(Bm + (long long)(k0 + bk) * ldb + n0 + bn + 8);
#ifndef HAVE_ASYNC_LDS
        uint4 av = make_uint4(0u, 0u, 0u, 0u);
        if (arow < M) av = *(const uint4*)(A + (long long)arow * lda + k0 + ac);
#endif
        if (k0 + 32 < K)
            __builtin_prefetch(Bm + (long long)(k0 + 32 + bk) * ldb + n0 + bn, 0, 1);
        __syncthreads();   // previous iteration's fragment reads done
#ifdef HAVE_ASYNC_LDS
        if (arow < M)
            __builtin_amdgcn_global_load_async_to_lds_b128(
                (AS1 v4i*)(A + (long long)arow * lda + k0 + ac),
                (AS3 v4i*)&As[ar * ASTR + ac], 0, 0);
#else
        {
            unsigned* asr = (unsigned*)&As[ar * ASTR + ac];
            asr[0] = av.x; asr[1] = av.y; asr[2] = av.z; asr[3] = av.w;
        }
#endif
        const u16 e[16] = { (u16)(bv0.x & 0xffffu), (u16)(bv0.x >> 16),
                            (u16)(bv0.y & 0xffffu), (u16)(bv0.y >> 16),
                            (u16)(bv0.z & 0xffffu), (u16)(bv0.z >> 16),
                            (u16)(bv0.w & 0xffffu), (u16)(bv0.w >> 16),
                            (u16)(bv1.x & 0xffffu), (u16)(bv1.x >> 16),
                            (u16)(bv1.y & 0xffffu), (u16)(bv1.y >> 16),
                            (u16)(bv1.z & 0xffffu), (u16)(bv1.z >> 16),
                            (u16)(bv1.w & 0xffffu), (u16)(bv1.w >> 16) };
#pragma unroll
        for (int j = 0; j < 16; ++j) Bs[(bn + j) * BSTR + bk] = e[j];
#ifdef HAVE_ASYNC_LDS
        __builtin_amdgcn_s_wait_asynccnt(0);
#endif
        __syncthreads();
        FragBF fa0, fa1, fb0, fb1;
        load_frag(As, ASTR, wm, lane, fa0);
        load_frag(As, ASTR, wm + 16, lane, fa1);
        load_frag(Bs, BSTR, wn, lane, fb0);
        load_frag(Bs, BSTR, wn + 16, lane, fb1);
        a00 = __builtin_amdgcn_wmma_f32_16x16x32_bf16(false, fa0.v, false, fb0.v, (short)0, a00, false, false);
        a01 = __builtin_amdgcn_wmma_f32_16x16x32_bf16(false, fa0.v, false, fb1.v, (short)0, a01, false, false);
        a10 = __builtin_amdgcn_wmma_f32_16x16x32_bf16(false, fa1.v, false, fb0.v, (short)0, a10, false, false);
        a11 = __builtin_amdgcn_wmma_f32_16x16x32_bf16(false, fa1.v, false, fb1.v, (short)0, a11, false, false);
    }
    const int gg = (lane >> 4) << 3;
    const int cn = lane & 15;
    const int c0 = n0 + wn + cn;
    const float b0 = bias ? bias[c0] : 0.f;
    const float b1 = bias ? bias[c0 + 16] : 0.f;
#pragma unroll
    for (int i = 0; i < 8; ++i) {
        const int mA = m0 + wm + gg + i;
        const int mB = mA + 16;
        if (mA < M) {
            C[(long long)mA * ldc + c0]      = a00[i] + b0;
            C[(long long)mA * ldc + c0 + 16] = a01[i] + b1;
        }
        if (mB < M) {
            C[(long long)mB * ldc + c0]      = a10[i] + b0;
            C[(long long)mB * ldc + c0 + 16] = a11[i] + b1;
        }
    }
}

// ---------------------------------------------------------------------------
// Prep / data movement kernels
// ---------------------------------------------------------------------------
__global__ void k_transpose_bf16(const float* __restrict__ in, u16* __restrict__ out,
                                 int R, int C) {               // out[C][R] = in[R][C]^T
    int i = blockIdx.x * 256 + threadIdx.x;
    if (i >= R * C) return;
    int r = i % R, c = i / R;
    out[i] = f2bf(in[r * C + c]);
}

__global__ void k_convert_bf16(const float* __restrict__ in, u16* __restrict__ out, int n) {
    int i = blockIdx.x * 256 + threadIdx.x;
    if (i < n) out[i] = f2bf(in[i]);
}

__global__ void k_build_fuse(const float* __restrict__ xir, const float* __restrict__ xlm,
                             u16* __restrict__ out) {          // (B*L0, 512) bf16, zero pad rows
    int i = blockIdx.x * 256 + threadIdx.x;
    if (i >= BB * L0 * 2 * CM) return;
    int j = i & 511; int p = i >> 9; int pos = p % L0; int b = p / L0;
    float v = 0.f;
    if (pos < NTOK) {
        if (j < CM) v = xir[((long long)b * NTOK + pos) * CM + j];
        else        v = xlm[((long long)b * NTOK + pos) * CM + (j - CM)];
    }
    out[i] = f2bf(v);
}

__global__ void k_zero_pad(float* __restrict__ xf) {           // zero rows pos in [NTOK,L0)
    int i = blockIdx.x * 256 + threadIdx.x;
    const int tot = BB * (L0 - NTOK) * CM;
    if (i >= tot) return;
    int c = i % CM; int p = i / CM;
    int pos = NTOK + p % (L0 - NTOK); int b = p / (L0 - NTOK);
    xf[((long long)b * L0 + pos) * CM + c] = 0.f;
}

__global__ void k_pool(const float* __restrict__ xf, u16* __restrict__ out, int Hs, int f) {
    const int Ls = Hs * Hs;
    int i = blockIdx.x * 256 + threadIdx.x;                    // (b, pos_s, c) c fastest
    if (i >= BB * Ls * CM) return;
    int c = i % CM; int p = i / CM; int ps = p % Ls; int b = p / Ls;
    int h0 = (ps / Hs) * f, w0 = (ps % Hs) * f;
    float s = 0.f;
    for (int a = 0; a < f; ++a)
        for (int e = 0; e < f; ++e)
            s += xf[((long long)b * L0 + (h0 + a) * H0 + (w0 + e)) * CM + c];
    out[i] = f2bf(s / (float)(f * f));
}

__global__ void k_conv_silu(const float* __restrict__ xz, const float* __restrict__ cw,
                            const float* __restrict__ cb, float* __restrict__ xc, int Hs) {
    const int Ls = Hs * Hs;
    int i = blockIdx.x * 256 + threadIdx.x;                    // (b,pos,d) d fastest
    if (i >= BB * Ls * DI) return;
    int d = i & (DI - 1); int p = i >> 9; int pos = p % Ls; int b = p / Ls;
    int h = pos / Hs, w = pos % Hs;
    float s = cb[d];
#pragma unroll
    for (int kh = 0; kh < 3; ++kh) {
        int hh = h + kh - 1; if (hh < 0 || hh >= Hs) continue;
#pragma unroll
        for (int kw = 0; kw < 3; ++kw) {
            int ww = w + kw - 1; if (ww < 0 || ww >= Hs) continue;
            s += xz[(long long)(b * Ls + hh * Hs + ww) * (2 * DI) + d] * cw[d * 9 + kh * 3 + kw];
        }
    }
    xc[i] = siluf(s);
}

__global__ void k_build_xs(const float* __restrict__ xc, u16* __restrict__ xs, int Hs) {
    const int Ls = Hs * Hs;
    int i = blockIdx.x * 256 + threadIdx.x;                    // ((b,k,d),l) l fastest
    if (i >= BB * KD * DI * Ls) return;
    int l = i % Ls; int r = i / Ls; int d = r % DI; int r2 = r / DI;
    int k = r2 & 3; int b = r2 >> 2;
    int src;
    if (k == 0)      src = l;
    else if (k == 1) { int w = l / Hs, h = l % Hs; src = h * Hs + w; }
    else if (k == 2) src = Ls - 1 - l;
    else { int l2 = Ls - 1 - l; int w = l2 / Hs, h = l2 % Hs; src = h * Hs + w; }
    xs[i] = f2bf(xc[(long long)(b * Ls + src) * DI + d]);
}

// ---------------------------------------------------------------------------
// Selective scan: one thread per (b,k,d) channel; dt_proj+softplus folded in.
// dts/Bs/Cs (48 x 64 chunk) and the u tile staged through LDS (async if avail).
// su row stride 68 u16 = 136B: 8B aligned for async b64, 34-bank -> conflict-free.
// ---------------------------------------------------------------------------
static constexpr int SUS = 68;

__global__ __launch_bounds__(256)
void k_scan(const float* __restrict__ xdbl, const u16* __restrict__ xs,
            const float* __restrict__ dtw_all, const float* __restrict__ dtb_all,
            const float* __restrict__ alog, const float* __restrict__ dsv,
            float* __restrict__ oy, int Ls)
{
    __shared__ float sdbl[48 * 64];
    __shared__ u16   su[256 * SUS];
    const int tid = threadIdx.x;
    const int d   = blockIdx.x * 256 + tid;
    const int z   = blockIdx.y;                 // z = b*4 + k
    const int k   = z & 3;
    const long long zdbl = (long long)z * 48 * Ls;
    const long long zxs  = (long long)z * DI * Ls;
    float dtw[RNK];
#pragma unroll
    for (int r = 0; r < RNK; ++r) dtw[r] = dtw_all[(k * DI + d) * RNK + r];
    const float dtb = dtb_all[k * DI + d];
    float Aa[NST];
#pragma unroll
    for (int n = 0; n < NST; ++n) Aa[n] = -__expf(alog[(k * DI + d) * NST + n]);
    const float Dd = dsv[k * DI + d];
    float h[NST];
#pragma unroll
    for (int n = 0; n < NST; ++n) h[n] = 0.f;
    const int d0 = blockIdx.x * 256;

    for (int l0 = 0; l0 < Ls; l0 += 64) {
#ifdef HAVE_ASYNC_LDS
        // sdbl: 48 rows x 64 f32 -> 16 b128 chunks per row (768 total)
        for (int ci = tid; ci < 768; ci += 256) {
            int row = ci >> 4, c4 = (ci & 15) << 2;  // float offset within row
            __builtin_amdgcn_global_load_async_to_lds_b128(
                (AS1 v4i*)(xdbl + zdbl + (long long)row * Ls + l0 + c4),
                (AS3 v4i*)&sdbl[(row << 6) + c4], 0, 0);
        }
        // su: 256 rows x 64 u16 -> 16 b64 chunks per row (4096 total)
        for (int ci = tid; ci < 4096; ci += 256) {
            int dd = ci >> 4, c4 = (ci & 15) << 2;   // u16 offset within row
            __builtin_amdgcn_global_load_async_to_lds_b64(
                (AS1 v2i*)(xs + zxs + (long long)(d0 + dd) * Ls + l0 + c4),
                (AS3 v2i*)&su[dd * SUS + c4], 0, 0);
        }
        __builtin_amdgcn_s_wait_asynccnt(0);
#else
        for (int i = tid; i < 48 * 64; i += 256)
            sdbl[i] = xdbl[zdbl + (long long)(i >> 6) * Ls + l0 + (i & 63)];
        for (int i = tid; i < 256 * 64; i += 256) {
            int dd = i >> 6, tt = i & 63;
            su[dd * SUS + tt] = xs[zxs + (long long)(d0 + dd) * Ls + l0 + tt];
        }
#endif
        __syncthreads();
        float* orow = oy + zxs + (long long)d * Ls + l0;
        for (int t = 0; t < 64; ++t) {
            float x = dtb;
#pragma unroll
            for (int r = 0; r < RNK; ++r) x += dtw[r] * sdbl[(r << 6) + t];
            const float delta = softplusf(x);
            const float u  = bf2f(su[tid * SUS + t]);
            const float du = delta * u;
            float y = 0.f;
#pragma unroll
            for (int n = 0; n < NST; ++n) {
                h[n] = h[n] * __expf(delta * Aa[n]) + du * sdbl[((16 + n) << 6) + t];
                y += h[n] * sdbl[((32 + n) << 6) + t];
            }
            orow[t] = y + Dd * u;
        }
        __syncthreads();
    }
}

__global__ void k_combine(const float* __restrict__ oy, float* __restrict__ ys, int Hs) {
    const int Ls = Hs * Hs;
    int i = blockIdx.x * 256 + threadIdx.x;                    // (b,l,d) d fastest
    if (i >= BB * Ls * DI) return;
    int d = i & (DI - 1); int p = i >> 9; int l = p % Ls; int b = p / Ls;
    int h = l / Hs, w = l % Hs;
    int lwh = w * Hs + h;
    long long base = ((long long)(b * 4) * DI + d) * Ls;
    long long sk   = (long long)DI * Ls;
    ys[i] = oy[base + l] + oy[base + 2 * sk + (Ls - 1 - l)]
          + oy[base + sk + lwh] + oy[base + 3 * sk + (Ls - 1 - lwh)];
}

__global__ __launch_bounds__(256)
void k_ln_gate(const float* __restrict__ ys, const float* __restrict__ xz,
               const float* __restrict__ wn, const float* __restrict__ bn,
               u16* __restrict__ yg)
{
    __shared__ float red[256];
    const int p = blockIdx.x;                                  // p = b*Ls + l
    const int tid = threadIdx.x;
    const float v0 = ys[(long long)p * DI + tid];
    const float v1 = ys[(long long)p * DI + 256 + tid];
    red[tid] = v0 + v1; __syncthreads();
    for (int o = 128; o > 0; o >>= 1) { if (tid < o) red[tid] += red[tid + o]; __syncthreads(); }
    const float mean = red[0] / (float)DI; __syncthreads();
    const float a0 = v0 - mean, a1 = v1 - mean;
    red[tid] = a0 * a0 + a1 * a1; __syncthreads();
    for (int o = 128; o > 0; o >>= 1) { if (tid < o) red[tid] += red[tid + o]; __syncthreads(); }
    const float rstd = rsqrtf(red[0] / (float)DI + 1e-5f);
    const float z0 = xz[(long long)p * (2 * DI) + DI + tid];
    const float z1 = xz[(long long)p * (2 * DI) + DI + 256 + tid];
    yg[(long long)p * DI + tid]       = f2bf((a0 * rstd * wn[tid] + bn[tid]) * siluf(z0));
    yg[(long long)p * DI + 256 + tid] = f2bf((a1 * rstd * wn[256 + tid] + bn[256 + tid]) * siluf(z1));
}

DEV float bilin(const float* __restrict__ y, int b, int Hs, int c, float fy, float fx) {
    const int Ls = Hs * Hs;
    int y0 = (int)floorf(fy), x0 = (int)floorf(fx);
    float wy = fy - (float)y0, wx = fx - (float)x0;
    int y0c = min(max(y0, 0), Hs - 1), y1c = min(max(y0 + 1, 0), Hs - 1);
    int x0c = min(max(x0, 0), Hs - 1), x1c = min(max(x0 + 1, 0), Hs - 1);
    const float* base = y + (long long)b * Ls * CM;
    float v00 = base[(y0c * Hs + x0c) * CM + c];
    float v01 = base[(y0c * Hs + x1c) * CM + c];
    float v10 = base[(y1c * Hs + x0c) * CM + c];
    float v11 = base[(y1c * Hs + x1c) * CM + c];
    return (1.f - wy) * ((1.f - wx) * v00 + wx * v01)
         + wy * ((1.f - wx) * v10 + wx * v11);
}

__global__ __launch_bounds__(256)
void k_final(const float* __restrict__ y1, const float* __restrict__ y2,
             const float* __restrict__ y3, const float* __restrict__ wn,
             const float* __restrict__ bnrm, float* __restrict__ out)
{
    __shared__ float red[256];
    const int p = blockIdx.x;                                  // b*NTOK + t
    const int b = p / NTOK, t = p % NTOK;
    const int tid = threadIdx.x;                               // channel
    const int h = t >> 6, w = t & 63;
    float v = y1[((long long)b * L0 + t) * CM + tid];
    v += bilin(y2, b, 32, tid, (h + 0.5f) * 0.5f - 0.5f,  (w + 0.5f) * 0.5f - 0.5f);
    v += bilin(y3, b, 16, tid, (h + 0.5f) * 0.25f - 0.5f, (w + 0.5f) * 0.25f - 0.5f);
    red[tid] = v; __syncthreads();
    for (int o = 128; o > 0; o >>= 1) { if (tid < o) red[tid] += red[tid + o]; __syncthreads(); }
    const float mean = red[0] / (float)CM; __syncthreads();
    const float a = v - mean;
    red[tid] = a * a; __syncthreads();
    for (int o = 128; o > 0; o >>= 1) { if (tid < o) red[tid] += red[tid + o]; __syncthreads(); }
    const float rstd = rsqrtf(red[0] / (float)CM + 1e-5f);
    out[(long long)p * CM + tid] = a * rstd * wn[tid] + bnrm[tid];
}

// ---------------------------------------------------------------------------
// Host orchestration
// ---------------------------------------------------------------------------
extern "C" void kernel_launch(void* const* d_in, const int* in_sizes, int n_in,
                              void* d_out, int out_size, void* d_ws, size_t ws_size,
                              hipStream_t stream)
{
    (void)in_sizes; (void)n_in; (void)out_size; (void)ws_size;
    const float* x_ir     = (const float*)d_in[0];
    const float* x_lm     = (const float*)d_in[1];
    const float* fusion_w = (const float*)d_in[2];
    const float* fusion_b = (const float*)d_in[3];
    const float* norm_w   = (const float*)d_in[37];
    const float* norm_b   = (const float*)d_in[38];

    size_t off = 0;
    auto alloc = [&](size_t bytes) -> void* {
        void* p = (char*)d_ws + off;
        off += (bytes + 255) & ~(size_t)255;
        return p;
    };
    float* xfused = (float*)alloc((size_t)BB * L0 * CM * 4);
    u16*   afuse  = (u16*)  alloc((size_t)BB * L0 * 2 * CM * 2);
    u16*   wTa    = (u16*)  alloc((size_t)CM * 2 * DI * 2);     // up to 256x1024
    u16*   wTo    = (u16*)  alloc((size_t)DI * CM * 2);
    u16*   wxp    = (u16*)  alloc((size_t)KD * 48 * DI * 2);
    u16*   xin    = (u16*)  alloc((size_t)BB * L0 * CM * 2);
    float* xz     = (float*)alloc((size_t)BB * L0 * 2 * DI * 4);
    float* xconv  = (float*)alloc((size_t)BB * L0 * DI * 4);
    u16*   xsb    = (u16*)  alloc((size_t)BB * KD * DI * L0 * 2);
    float* xdbl   = (float*)alloc((size_t)BB * KD * 48 * L0 * 4);
    float* oyb    = (float*)alloc((size_t)BB * KD * DI * L0 * 4);
    float* ysum   = (float*)alloc((size_t)BB * L0 * DI * 4);
    u16*   ygate  = (u16*)  alloc((size_t)BB * L0 * DI * 2);
    float* yscale[3];
    yscale[0] = (float*)alloc((size_t)BB * L0 * CM * 4);
    yscale[1] = (float*)alloc((size_t)BB * 1024 * CM * 4);
    yscale[2] = (float*)alloc((size_t)BB * 256 * CM * 4);

    auto cdiv = [](long long a, long long b) { return (unsigned)((a + b - 1) / b); };

    // ---- fusion: x = concat(x_ir,x_lm) @ fusion_w^T + b, pad to 64x64 grid ----
    k_build_fuse<<<cdiv((long long)BB * L0 * 2 * CM, 256), 256, 0, stream>>>(x_ir, x_lm, afuse);
    k_transpose_bf16<<<cdiv(CM * 2 * CM, 256), 256, 0, stream>>>(fusion_w, wTa, CM, 2 * CM);
    {
        dim3 g(CM / 128, (BB * L0) / 64, 1);
        gemm_bf16_kernel<<<g, 256, 0, stream>>>(afuse, wTa, xfused, fusion_b,
            BB * L0, CM, 2 * CM, 2 * CM, CM, CM, 0, 0, 0, 1);
    }
    k_zero_pad<<<cdiv((long long)BB * (L0 - NTOK) * CM, 256), 256, 0, stream>>>(xfused);

    // ---- per-scale SS2D ----
    for (int s = 0; s < 3; ++s) {
        const int Hs = H0 >> s, Ls = Hs * Hs, f = 1 << s;
        const int pb = 4 + 11 * s;
        const float* inw  = (const float*)d_in[pb + 0];
        const float* cwv  = (const float*)d_in[pb + 1];
        const float* cbv  = (const float*)d_in[pb + 2];
        const float* xpw  = (const float*)d_in[pb + 3];
        const float* dtpw = (const float*)d_in[pb + 4];
        const float* dtpb = (const float*)d_in[pb + 5];
        const float* alog = (const float*)d_in[pb + 6];
        const float* dsv  = (const float*)d_in[pb + 7];
        const float* onw  = (const float*)d_in[pb + 8];
        const float* onb  = (const float*)d_in[pb + 9];
        const float* opw  = (const float*)d_in[pb + 10];

        k_pool<<<cdiv((long long)BB * Ls * CM, 256), 256, 0, stream>>>(xfused, xin, Hs, f);

        // in_proj: (B*Ls,256) @ (256,1024)
        k_transpose_bf16<<<cdiv(2 * DI * CM, 256), 256, 0, stream>>>(inw, wTa, 2 * DI, CM);
        {
            dim3 g((2 * DI) / 128, (BB * Ls) / 64, 1);
            gemm_bf16_kernel<<<g, 256, 0, stream>>>(xin, wTa, xz, nullptr,
                BB * Ls, 2 * DI, CM, CM, 2 * DI, 2 * DI, 0, 0, 0, 1);
        }
        k_conv_silu<<<cdiv((long long)BB * Ls * DI, 256), 256, 0, stream>>>(xz, cwv, cbv, xconv, Hs);
        k_build_xs<<<cdiv((long long)BB * KD * DI * Ls, 256), 256, 0, stream>>>(xconv, xsb, Hs);

        // x_proj: per (b,k): (48,512) @ (512,Ls) -> (48,Ls)
        k_convert_bf16<<<cdiv(KD * 48 * DI, 256), 256, 0, stream>>>(xpw, wxp, KD * 48 * DI);
        {
            dim3 g(Ls / 128, 1, BB * KD);
            gemm_bf16_kernel<<<g, 256, 0, stream>>>(wxp, xsb, xdbl, nullptr,
                48, Ls, DI, DI, Ls, Ls,
                (long long)48 * DI, (long long)DI * Ls, (long long)48 * Ls, KD);
        }
        // selective scan (dt_proj folded in)
        {
            dim3 g(DI / 256, BB * KD);
            k_scan<<<g, 256, 0, stream>>>(xdbl, xsb, dtpw, dtpb, alog, dsv, oyb, Ls);
        }
        k_combine<<<cdiv((long long)BB * Ls * DI, 256), 256, 0, stream>>>(oyb, ysum, Hs);
        k_ln_gate<<<BB * Ls, 256, 0, stream>>>(ysum, xz, onw, onb, ygate);

        // out_proj: (B*Ls,512) @ (512,256)
        k_transpose_bf16<<<cdiv(CM * DI, 256), 256, 0, stream>>>(opw, wTo, CM, DI);
        {
            dim3 g(CM / 128, (BB * Ls) / 64, 1);
            gemm_bf16_kernel<<<g, 256, 0, stream>>>(ygate, wTo, yscale[s], nullptr,
                BB * Ls, CM, DI, DI, CM, CM, 0, 0, 0, 1);
        }
    }

    // ---- pyramid fuse (bilinear up) + final LN ----
    k_final<<<BB * NTOK, 256, 0, stream>>>(yscale[0], yscale[1], yscale[2],
                                           norm_w, norm_b, (float*)d_out);
}
